// SABlock_82291573392018
// MI455X (gfx1250) — matrix operand order repaired and checked by reading, not compiled, not run on hardware
//
#include <hip/hip_runtime.h>
#include <hip/hip_bf16.h>
#include <stdint.h>

// ---------------------------------------------------------------------------
// ViT self-attention block for MI455X (gfx1250, wave32, WMMA 16x16x32 f16).
//   out = softmax((xWq)(xWk)^T * d^-0.5) (xWv) @ W_out + b_out
// All GEMMs run on v_wmma_f32_16x16x32_f16 with fp32 accumulators.
// Fragments are loaded directly as paired global_load_b128 (A: row-major rows,
// B: K-contiguous transposed operands), no LDS staging needed except for the
// P-matrix C->A layout conversion in flash attention.
// ---------------------------------------------------------------------------

typedef __attribute__((ext_vector_type(16))) _Float16 v16h;
typedef __attribute__((ext_vector_type(8)))  float    v8f;
typedef _Float16 f16;

constexpr int kDim  = 768;
constexpr int kNH   = 12;
constexpr int kHD   = 64;
constexpr int kB    = 8;
constexpr int kN    = 1024;
constexpr int kM    = kB * kN;      // 8192 rows
constexpr int kQKV  = 3 * kDim;     // 2304

static __device__ __forceinline__ v8f wmma_f16(v16h a, v16h b, v8f c) {
  return __builtin_amdgcn_wmma_f32_16x16x32_f16(false, a, false, b, (short)0, c,
                                                false, false);
}

// CDNA5 split-counter wait: LDS ops complete (also a compiler barrier).
static __device__ __forceinline__ void wait_dscnt0() {
  asm volatile("s_wait_dscnt 0" ::: "memory");
}

// -------------------------- conversion kernels -----------------------------

__global__ __launch_bounds__(256) void cvt_f32_to_f16(const float* __restrict__ src,
                                                      f16* __restrict__ dst, int n) {
  int i = blockIdx.x * blockDim.x + threadIdx.x;
  int stride = gridDim.x * blockDim.x;
  for (; i < n; i += stride) dst[i] = (f16)src[i];
}

// W: [rows][cols] fp32  ->  Wt: [cols][rows] f16  (so K-dim is contiguous)
__global__ __launch_bounds__(256) void transpose_cvt(const float* __restrict__ W,
                                                     f16* __restrict__ Wt,
                                                     int rows, int cols) {
  int i = blockIdx.x * blockDim.x + threadIdx.x;
  int total = rows * cols;
  int stride = gridDim.x * blockDim.x;
  for (; i < total; i += stride) {
    int r = i / cols, c = i - r * cols;
    Wt[(size_t)c * rows + r] = (f16)W[i];
  }
}

// ------------------------------ QKV GEMM -----------------------------------
// C[8192][2304] = xh[8192][768] @ Wqkv ;  B supplied transposed f16 [2304][768]
// Block: 128 thr (4 waves). Block tile 128x64; each wave: 32 rows x 64 cols
// (acc[2][4]) -> 8 WMMAs per K-step against 12 b128 loads.
// n0 is 64-aligned so the whole block maps to exactly one of Q/K/V and one
// head band: `which`, `h`, batch index are block-uniform (no divergence).

__global__ __launch_bounds__(128) void qkv_kernel(const f16* __restrict__ xh,
                                                  const f16* __restrict__ wqkvT,
                                                  f16* __restrict__ Qh,
                                                  f16* __restrict__ Kh,
                                                  f16* __restrict__ Vt) {
  const int lane = threadIdx.x & 31;
  const int wave = threadIdx.x >> 5;
  const int lrow = lane & 15;
  const int hi   = (lane & 16) ? 1 : 0;
  const int n0 = blockIdx.x * 64;
  const int m0 = blockIdx.y * 128 + wave * 32;

  v8f acc[2][4] = {};

  const uint32_t* xrow0 = (const uint32_t*)(xh + (size_t)(m0 + lrow) * kDim);
  const uint32_t* xrow1 = (const uint32_t*)(xh + (size_t)(m0 + 16 + lrow) * kDim);
  for (int kt = 0; kt < kDim; kt += 32) {
    v16h a0, a1;
    uint32_t* ap0 = (uint32_t*)&a0;
    uint32_t* ap1 = (uint32_t*)&a1;
#pragma unroll
    for (int r = 0; r < 8; ++r) {
      // A layout: k = kt + (hi?16:0) + 2r
      int idx = (kt >> 1) + (hi ? 8 : 0) + r;
      ap0[r] = xrow0[idx];
      ap1[r] = xrow1[idx];
    }
#pragma unroll
    for (int s = 0; s < 4; ++s) {
      const uint32_t* wrow =
          (const uint32_t*)(wqkvT + (size_t)(n0 + s * 16 + lrow) * kDim);
      v16h b; uint32_t* bp = (uint32_t*)&b;
#pragma unroll
      for (int r = 0; r < 8; ++r) {
        // B layout: k = kt + (hi?16:0) + 2r
        bp[r] = wrow[(kt >> 1) + (hi ? 8 : 0) + r];
      }
      acc[0][s] = wmma_f16(a0, b, acc[0][s]);
      acc[1][s] = wmma_f16(a1, b, acc[1][s]);
    }
  }

  // ---- epilogue: all routing indices are block-uniform ----
  const int which = n0 / kDim;                 // 0=Q, 1=K, 2=V
  const int h     = (n0 - which * kDim) >> 6;  // head band (64-aligned)
  const int bb    = m0 >> 10;                  // 128-row tile within one batch
  const int tok0  = m0 & (kN - 1);
  const size_t bhI = (size_t)(bb * kNH + h);

#pragma unroll
  for (int ra = 0; ra < 2; ++ra) {
#pragma unroll
    for (int s = 0; s < 4; ++s) {
      int d = s * 16 + lrow;
#pragma unroll
      for (int j = 0; j < 8; ++j) {
        int tok = tok0 + ra * 16 + j + hi * 8;   // C layout: row j + 8*hi
        f16 v = (f16)acc[ra][s][j];
        if (which == 0)      Qh[(bhI * kN + tok) * kHD + d] = v;
        else if (which == 1) Kh[(bhI * kN + tok) * kHD + d] = v;
        else                 Vt[(bhI * kHD + d) * kN + tok] = v;  // transposed
      }
    }
  }
}

// --------------------------- flash attention -------------------------------
// Block 256 thr (8 waves) per (b,h, 128-query chunk). Wave owns 16 q-rows,
// streams 32-key tiles: S = Q K^T (2 subtiles x 2 K-chunks), online softmax,
// P->LDS->A-fragment, O += P V using transposed V (direct global B-fragments).

__global__ __launch_bounds__(256) void attn_kernel(const f16* __restrict__ Qh,
                                                   const f16* __restrict__ Kh,
                                                   const f16* __restrict__ Vt,
                                                   f16* __restrict__ attn) {
  __shared__ f16 Ps[8][16][34];   // per-wave 16x32 P tile (pitch 34 halves)

  const int lane = threadIdx.x & 31;
  const int wave = threadIdx.x >> 5;
  const int lrow = lane & 15;
  const int hi   = (lane & 16) ? 1 : 0;
  const int bh = blockIdx.y;
  const int bb = bh / kNH;
  const int hh = bh - bb * kNH;
  const int q0 = blockIdx.x * 128 + wave * 16;

  const f16* Qbh = Qh + (size_t)bh * kN * kHD;
  const f16* Kbh = Kh + (size_t)bh * kN * kHD;
  const f16* Vbh = Vt + (size_t)bh * kHD * kN;

  // Q fragments (d = 0..31 and 32..63), kept in registers for the whole loop.
  v16h aq0, aq1;
  {
    const uint32_t* qrow = (const uint32_t*)(Qbh + (size_t)(q0 + lrow) * kHD);
    uint32_t* a0 = (uint32_t*)&aq0; uint32_t* a1 = (uint32_t*)&aq1;
#pragma unroll
    for (int r = 0; r < 8; ++r) {
      int idx = (hi ? 8 : 0) + r;
      a0[r] = qrow[idx];
      a1[r] = qrow[idx + 16];
    }
  }

  v8f o[4] = {};
  float Mj[8], Lj[8];
#pragma unroll
  for (int j = 0; j < 8; ++j) { Mj[j] = -1e30f; Lj[j] = 0.f; }

  const float scale = 0.125f;   // 64^-0.5

  for (int kt0 = 0; kt0 < kN; kt0 += 32) {
    // ---- S = Q @ K^T for 32 keys (two 16-wide subtiles) ----
    v8f sAcc[2] = {};
#pragma unroll
    for (int sub = 0; sub < 2; ++sub) {
      // B operand: k-dim = head dim (contiguous in K storage) -> direct loads
      const uint32_t* krow =
          (const uint32_t*)(Kbh + (size_t)(kt0 + sub * 16 + lrow) * kHD);
      v16h b0, b1; uint32_t* p0 = (uint32_t*)&b0; uint32_t* p1 = (uint32_t*)&b1;
#pragma unroll
      for (int r = 0; r < 8; ++r) {
        int u = (hi ? 8 : 0) + r;       // dword index: d = (hi?16:0) + 2r
        p0[r] = krow[u];
        p1[r] = krow[u + 16];           // d + 32
      }
      sAcc[sub] = wmma_f16(aq0, b0, sAcc[sub]);
      sAcc[sub] = wmma_f16(aq1, b1, sAcc[sub]);
    }

    // ---- online softmax (row stats per C-layout row: reg j, lane half) ----
#pragma unroll
    for (int j = 0; j < 8; ++j) {
      float v0 = sAcc[0][j] * scale;
      float v1 = sAcc[1][j] * scale;
      float mx = fmaxf(v0, v1);
#pragma unroll
      for (int off = 1; off < 16; off <<= 1)
        mx = fmaxf(mx, __shfl_xor(mx, off, 32));   // stays inside 16-lane half
      float newM = fmaxf(Mj[j], mx);
      float corr = __expf(Mj[j] - newM);
      Mj[j] = newM;
      float p0 = __expf(v0 - newM);
      float p1 = __expf(v1 - newM);
      float ps = p0 + p1;
#pragma unroll
      for (int off = 1; off < 16; off <<= 1)
        ps += __shfl_xor(ps, off, 32);
      Lj[j] = Lj[j] * corr + ps;
#pragma unroll
      for (int s = 0; s < 4; ++s) o[s][j] *= corr;
      int row = j + hi * 8;
      Ps[wave][row][lrow]      = (f16)p0;
      Ps[wave][row][16 + lrow] = (f16)p1;
    }
    wait_dscnt0();   // P stores visible before A-frag loads (same wave)

    // ---- P as A-fragment (from per-wave LDS tile) ----
    v16h pa; uint32_t* pp = (uint32_t*)&pa;
    const uint32_t* prow = (const uint32_t*)&Ps[wave][lrow][0];
#pragma unroll
    for (int r = 0; r < 8; ++r) {
      int k = (hi ? 16 : 0) + r * 2;
      pp[r] = prow[k >> 1];
    }

    // ---- O += P @ V  (V transposed: B-fragment = direct dword loads) ----
#pragma unroll
    for (int s = 0; s < 4; ++s) {
      const uint32_t* vrow =
          (const uint32_t*)(Vbh + (size_t)(s * 16 + lrow) * kN + kt0);
      v16h bv; uint32_t* bp = (uint32_t*)&bv;
#pragma unroll
      for (int r = 0; r < 8; ++r)
        bp[r] = vrow[(hi ? 8 : 0) + r];   // tok = kt0 + (hi?16:0) + 2r
      o[s] = wmma_f16(pa, bv, o[s]);
    }
  }

  // ---- normalize and write attn output [b][tok][h*64+d] in f16 ----
#pragma unroll
  for (int j = 0; j < 8; ++j) {
    float inv = 1.0f / Lj[j];
    int tok = q0 + j + hi * 8;
    size_t base = ((size_t)bb * kN + tok) * kDim + hh * kHD;
#pragma unroll
    for (int s = 0; s < 4; ++s)
      attn[base + s * 16 + lrow] = (f16)(o[s][j] * inv);
  }
}

// --------------------------- output projection -----------------------------
// out[8192][768] = attn[8192][768] @ W_out + b_out ; W supplied as f16 [n][k].
// Same 128x64 block tile / 32-row waves as qkv_kernel.

__global__ __launch_bounds__(128) void proj_kernel(const f16* __restrict__ attn,
                                                   const f16* __restrict__ woutT,
                                                   const float* __restrict__ bias,
                                                   float* __restrict__ out) {
  const int lane = threadIdx.x & 31;
  const int wave = threadIdx.x >> 5;
  const int lrow = lane & 15;
  const int hi   = (lane & 16) ? 1 : 0;
  const int n0 = blockIdx.x * 64;
  const int m0 = blockIdx.y * 128 + wave * 32;

  v8f acc[2][4] = {};

  const uint32_t* arow0 = (const uint32_t*)(attn + (size_t)(m0 + lrow) * kDim);
  const uint32_t* arow1 = (const uint32_t*)(attn + (size_t)(m0 + 16 + lrow) * kDim);
  for (int kt = 0; kt < kDim; kt += 32) {
    v16h a0, a1;
    uint32_t* ap0 = (uint32_t*)&a0;
    uint32_t* ap1 = (uint32_t*)&a1;
#pragma unroll
    for (int r = 0; r < 8; ++r) {
      int idx = (kt >> 1) + (hi ? 8 : 0) + r;
      ap0[r] = arow0[idx];
      ap1[r] = arow1[idx];
    }
#pragma unroll
    for (int s = 0; s < 4; ++s) {
      const uint32_t* wrow =
          (const uint32_t*)(woutT + (size_t)(n0 + s * 16 + lrow) * kDim);
      v16h b; uint32_t* bp = (uint32_t*)&b;
#pragma unroll
      for (int r = 0; r < 8; ++r)
        bp[r] = wrow[(kt >> 1) + (hi ? 8 : 0) + r];
      acc[0][s] = wmma_f16(a0, b, acc[0][s]);
      acc[1][s] = wmma_f16(a1, b, acc[1][s]);
    }
  }

#pragma unroll
  for (int s = 0; s < 4; ++s) {
    int c = n0 + s * 16 + lrow;
    float bv = bias[c];
#pragma unroll
    for (int ra = 0; ra < 2; ++ra) {
#pragma unroll
      for (int j = 0; j < 8; ++j) {
        int m = m0 + ra * 16 + j + hi * 8;
        out[(size_t)m * kDim + c] = acc[ra][s][j] + bv;
      }
    }
  }
}

// ------------------------------- launcher ----------------------------------

extern "C" void kernel_launch(void* const* d_in, const int* in_sizes, int n_in,
                              void* d_out, int out_size, void* d_ws, size_t ws_size,
                              hipStream_t stream) {
  const float* x     = (const float*)d_in[0];   // [8,1024,768]
  const float* W_qkv = (const float*)d_in[1];   // [768,2304]
  const float* W_out = (const float*)d_in[2];   // [768,768]
  const float* b_out = (const float*)d_in[3];   // [768]
  float* out = (float*)d_out;

  char* ws = (char*)d_ws;
  size_t off = 0;
  auto alloc = [&](size_t bytes) -> void* {
    void* p = ws + off;
    off += (bytes + 255) & ~(size_t)255;
    return p;
  };

  f16* xh    = (f16*)alloc((size_t)kM * kDim * sizeof(f16));     // also reused as attn
  f16* wqkvT = (f16*)alloc((size_t)kQKV * kDim * sizeof(f16));
  f16* woutT = (f16*)alloc((size_t)kDim * kDim * sizeof(f16));
  f16* Qh    = (f16*)alloc((size_t)kB * kNH * kN * kHD * sizeof(f16));
  f16* Kh    = (f16*)alloc((size_t)kB * kNH * kN * kHD * sizeof(f16));
  f16* Vt    = (f16*)alloc((size_t)kB * kNH * kHD * kN * sizeof(f16));
  f16* attn  = xh;   // xh is dead after qkv_kernel; alias to save workspace

  cvt_f32_to_f16<<<1024, 256, 0, stream>>>(x, xh, kM * kDim);
  transpose_cvt<<<1024, 256, 0, stream>>>(W_qkv, wqkvT, kDim, kQKV);
  transpose_cvt<<<512, 256, 0, stream>>>(W_out, woutT, kDim, kDim);

  qkv_kernel<<<dim3(kQKV / 64, kM / 128), 128, 0, stream>>>(xh, wqkvT, Qh, Kh, Vt);
  attn_kernel<<<dim3(kN / 128, kB * kNH), 256, 0, stream>>>(Qh, Kh, Vt, attn);
  proj_kernel<<<dim3(kDim / 64, kM / 128), 128, 0, stream>>>(attn, woutT, b_out, out);
}
